// ModifiedLinear_24309514895727
// MI455X (gfx1250) — matrix operand order repaired and checked by reading, not compiled, not run on hardware
//
#include <hip/hip_runtime.h>

// ---------------------------------------------------------------------------
// Problem constants (from reference): P=64, IN=1024, OUT=1024, M = 8*2048.
// out = x @ (weight + sum_i scale[i]*wpert[i]).T + (bias + sum_i scale[i]*bpert[i])
//
// Strategy (MI455X / gfx1250, wave32):
//  * Fold perturbations once into w_eff; split w_eff and x into fp16 hi+lo
//    halves and run the GEMM with V_WMMA_F32_16X16X32_F16 using the
//    3-product split-precision scheme (near-fp32 accuracy, ~8x the MAC rate
//    of the only native f32 WMMA shape 16x16x4).
//  * x (64MB) and w_eff (4MB) fit in the 192MB L2, so the kernel is close to
//    the ~11us HBM roofline (256MB total traffic at 23.3 TB/s).
// ---------------------------------------------------------------------------

typedef __attribute__((ext_vector_type(4)))  _Float16 v4h;
typedef __attribute__((ext_vector_type(8)))  _Float16 v8h;
typedef __attribute__((ext_vector_type(16))) _Float16 v16h;
typedef __attribute__((ext_vector_type(8)))  float    v8f;

#define P_DIM   64
#define IN_DIM  1024
#define OUT_DIM 1024

#define BM 128
#define BN 128
#define BK 32
#define LDS_STRIDE 40   // 32 halfs + 8 pad (80B rows; 16B-aligned, conflict-spreading)

// ---------------------------------------------------------------------------
// Kernel 1: w_eff = weight + sum_i scale[i]*wpert[i]; split into fp16 hi/lo.
// 256 threads/block, each thread owns 8 contiguous elements (16B fp16 loads).
// ---------------------------------------------------------------------------
__global__ __launch_bounds__(256) void fuse_weight_kernel(
    const float* __restrict__ weight,
    const float* __restrict__ scale,
    const _Float16* __restrict__ wpert,
    _Float16* __restrict__ w_hi,
    _Float16* __restrict__ w_lo) {
  __shared__ float s_scale[P_DIM];
  const int t = threadIdx.x;
  if (t < P_DIM) s_scale[t] = scale[t];
  __syncthreads();

  const size_t base = ((size_t)blockIdx.x * blockDim.x + t) * 8;

  float acc[8];
  {
    const float4 w0 = *(const float4*)(weight + base);
    const float4 w1 = *(const float4*)(weight + base + 4);
    acc[0] = w0.x; acc[1] = w0.y; acc[2] = w0.z; acc[3] = w0.w;
    acc[4] = w1.x; acc[5] = w1.y; acc[6] = w1.z; acc[7] = w1.w;
  }

  #pragma unroll 4
  for (int i = 0; i < P_DIM; ++i) {
    const v8h pv = *(const v8h*)(wpert + (size_t)i * (OUT_DIM * IN_DIM) + base);
    const float s = s_scale[i];
    #pragma unroll
    for (int e = 0; e < 8; ++e) acc[e] = fmaf(s, (float)pv[e], acc[e]);
  }

  v8h hi, lo;
  #pragma unroll
  for (int e = 0; e < 8; ++e) {
    const _Float16 h = (_Float16)acc[e];
    hi[e] = h;
    lo[e] = (_Float16)(acc[e] - (float)h);
  }
  *(v8h*)(w_hi + base) = hi;
  *(v8h*)(w_lo + base) = lo;
}

// ---------------------------------------------------------------------------
// Kernel 2: b_eff = bias + sum_i scale[i]*bpert[i]  (1024 outputs, tiny)
// ---------------------------------------------------------------------------
__global__ __launch_bounds__(256) void fuse_bias_kernel(
    const float* __restrict__ bias,
    const float* __restrict__ scale,
    const _Float16* __restrict__ bpert,
    float* __restrict__ b_eff) {
  const int j = blockIdx.x * blockDim.x + threadIdx.x;
  float acc = bias[j];
  #pragma unroll 4
  for (int i = 0; i < P_DIM; ++i)
    acc = fmaf(scale[i], (float)bpert[(size_t)i * OUT_DIM + j], acc);
  b_eff[j] = acc;
}

// ---------------------------------------------------------------------------
// Kernel 3: GEMM  out[m,n] = sum_k x[m,k]*w_eff[n,k] + b_eff[n]
// Block: 256 threads = 8 waves; 128x128 output tile; BK=32.
// Wave (wr,wc): wr in 0..3 (M strips of 32), wc in 0..1 (N strips of 64).
// Each wave: 2 (M) x 4 (N) tiles of 16x16, v8f accumulators, 24 WMMAs/K-step.
//
// Fragment layouts per CDNA5 ISA 7.12.2 (wave32):
//  A 16x32 f16:  lane L -> row M=L%16; halfs 0..7 = K[kb..kb+7],
//                halfs 8..15 = K[16+kb..16+kb+7], kb = (L/16)*8.
//  B 32x16 f16:  lane L -> col N=L%16; halfs 0..7 = K[kb..kb+7], 8..15 = K[16+kb..16+kb+7] (as A).
//  C/D 16x16 f32: lane L -> N=L%16, vgpr r -> M = r + 8*(L/16).
// ---------------------------------------------------------------------------
__global__ __launch_bounds__(256) void gemm_split_f16_kernel(
    const float* __restrict__ x,
    const _Float16* __restrict__ w_hi,
    const _Float16* __restrict__ w_lo,
    const float* __restrict__ b_eff,
    float* __restrict__ out) {
  __shared__ _Float16 sA_hi[BM * LDS_STRIDE];
  __shared__ _Float16 sA_lo[BM * LDS_STRIDE];
  __shared__ _Float16 sB_hi[BN * LDS_STRIDE];
  __shared__ _Float16 sB_lo[BN * LDS_STRIDE];

  const int t    = threadIdx.x;
  const int lane = t & 31;
  const int wave = t >> 5;      // 0..7
  const int wr   = wave >> 1;   // 0..3  -> M offset wr*32
  const int wc   = wave & 1;    // 0..1  -> N offset wc*64

  const int block_m = blockIdx.y * BM;
  const int block_n = blockIdx.x * BN;

  v8f acc[2][4];
  #pragma unroll
  for (int mt = 0; mt < 2; ++mt)
    #pragma unroll
    for (int nt = 0; nt < 4; ++nt)
      acc[mt][nt] = (v8f)(0.0f);

  // Staging thread mappings:
  //  x tile: 128 rows x 32 f32 = 1024 float4; thread t does 4 rows (p*32 strides)
  const int xrow = t >> 3;              // 0..31
  const int xcol = (t & 7) * 4;         // 0,4,...,28
  //  w tiles: 128 rows x 32 f16 = 512 v8h per plane; thread t does 2 rows
  const int wrow   = t >> 2;            // 0..63
  const int wchunk = (t & 3) * 8;       // 0,8,16,24

  const int nlane = lane & 15;
  const int kbase = (lane >> 4) * 8;    // A: 0 or 8
  const int krun  = (lane >> 4) * 16;   // unused; B uses kbase too

  for (int k0 = 0; k0 < IN_DIM; k0 += BK) {
    // ---- stage x tile (convert fp32 -> hi/lo fp16) ----
    #pragma unroll
    for (int p = 0; p < 4; ++p) {
      const int r = xrow + p * 32;
      const float4 v =
          *(const float4*)(x + (size_t)(block_m + r) * IN_DIM + k0 + xcol);
      const float vs[4] = {v.x, v.y, v.z, v.w};
      v4h h4, l4;
      #pragma unroll
      for (int e = 0; e < 4; ++e) {
        const _Float16 h = (_Float16)vs[e];
        h4[e] = h;
        l4[e] = (_Float16)(vs[e] - (float)h);
      }
      *(v4h*)&sA_hi[r * LDS_STRIDE + xcol] = h4;
      *(v4h*)&sA_lo[r * LDS_STRIDE + xcol] = l4;
    }
    // ---- stage w tiles (already fp16 hi/lo in workspace) ----
    #pragma unroll
    for (int p = 0; p < 2; ++p) {
      const int r = wrow + p * 64;
      const size_t g = (size_t)(block_n + r) * IN_DIM + k0 + wchunk;
      const v8h h = *(const v8h*)(w_hi + g);
      const v8h l = *(const v8h*)(w_lo + g);
      *(v8h*)&sB_hi[r * LDS_STRIDE + wchunk] = h;
      *(v8h*)&sB_lo[r * LDS_STRIDE + wchunk] = l;
    }
    __syncthreads();

    // ---- load A fragments (2 M-tiles, hi+lo) ----
    v16h a_hi[2], a_lo[2];
    #pragma unroll
    for (int mt = 0; mt < 2; ++mt) {
      const int arow = wr * 32 + mt * 16 + nlane;
      {
        const _Float16* pa = &sA_hi[arow * LDS_STRIDE];
        const v8h a0 = *(const v8h*)(pa + kbase);
        const v8h a1 = *(const v8h*)(pa + kbase + 16);
        a_hi[mt] = __builtin_shufflevector(a0, a1, 0, 1, 2, 3, 4, 5, 6, 7, 8,
                                           9, 10, 11, 12, 13, 14, 15);
      }
      {
        const _Float16* pa = &sA_lo[arow * LDS_STRIDE];
        const v8h a0 = *(const v8h*)(pa + kbase);
        const v8h a1 = *(const v8h*)(pa + kbase + 16);
        a_lo[mt] = __builtin_shufflevector(a0, a1, 0, 1, 2, 3, 4, 5, 6, 7, 8,
                                           9, 10, 11, 12, 13, 14, 15);
      }
    }

    // ---- B fragments + WMMA (3-product split accumulation) ----
    #pragma unroll
    for (int nt = 0; nt < 4; ++nt) {
      const int brow = wc * 64 + nt * 16 + nlane;
      v16h b_hi, b_lo;
      {
        const _Float16* pb = &sB_hi[brow * LDS_STRIDE + kbase];
        const v8h b0 = *(const v8h*)pb;
        const v8h b1 = *(const v8h*)(pb + 16);
        b_hi = __builtin_shufflevector(b0, b1, 0, 1, 2, 3, 4, 5, 6, 7, 8, 9,
                                       10, 11, 12, 13, 14, 15);
      }
      {
        const _Float16* pb = &sB_lo[brow * LDS_STRIDE + kbase];
        const v8h b0 = *(const v8h*)pb;
        const v8h b1 = *(const v8h*)(pb + 16);
        b_lo = __builtin_shufflevector(b0, b1, 0, 1, 2, 3, 4, 5, 6, 7, 8, 9,
                                       10, 11, 12, 13, 14, 15);
      }
      #pragma unroll
      for (int mt = 0; mt < 2; ++mt) {
        acc[mt][nt] = __builtin_amdgcn_wmma_f32_16x16x32_f16(
            false, a_hi[mt], false, b_hi, (short)0, acc[mt][nt], false, false);
        acc[mt][nt] = __builtin_amdgcn_wmma_f32_16x16x32_f16(
            false, a_hi[mt], false, b_lo, (short)0, acc[mt][nt], false, false);
        acc[mt][nt] = __builtin_amdgcn_wmma_f32_16x16x32_f16(
            false, a_lo[mt], false, b_hi, (short)0, acc[mt][nt], false, false);
      }
    }
    __syncthreads();
  }

  // ---- epilogue: add bias, store fp32 ----
  const int lhalf = lane >> 4;  // 0 -> rows 0..7, 1 -> rows 8..15 of the tile
  #pragma unroll
  for (int nt = 0; nt < 4; ++nt) {
    const int n = block_n + wc * 64 + nt * 16 + nlane;
    const float b = b_eff[n];
    #pragma unroll
    for (int mt = 0; mt < 2; ++mt) {
      const int m0 = block_m + wr * 32 + mt * 16 + lhalf * 8;
      #pragma unroll
      for (int r = 0; r < 8; ++r)
        out[(size_t)(m0 + r) * OUT_DIM + n] = acc[mt][nt][r] + b;
    }
  }
}

// ---------------------------------------------------------------------------
// Launch: inputs in setup_inputs() order:
//  0: x (M*IN f32)  1: weight (OUT*IN f32)  2: bias (OUT f32)
//  3: scale (P f32) 4: wpert (P*OUT*IN f16) 5: bpert (P*OUT f16)
// Workspace layout: [0,2MB) w_hi f16 | [2MB,4MB) w_lo f16 | [4MB,+4KB) b_eff f32
// ---------------------------------------------------------------------------
extern "C" void kernel_launch(void* const* d_in, const int* in_sizes, int n_in,
                              void* d_out, int out_size, void* d_ws,
                              size_t ws_size, hipStream_t stream) {
  const float*    x      = (const float*)d_in[0];
  const float*    weight = (const float*)d_in[1];
  const float*    bias   = (const float*)d_in[2];
  const float*    scale  = (const float*)d_in[3];
  const _Float16* wpert  = (const _Float16*)d_in[4];
  const _Float16* bpert  = (const _Float16*)d_in[5];
  float*          out    = (float*)d_out;

  char* ws = (char*)d_ws;
  _Float16* w_hi  = (_Float16*)ws;
  _Float16* w_lo  = (_Float16*)(ws + (size_t)OUT_DIM * IN_DIM * sizeof(_Float16));
  float*    b_eff = (float*)(ws + (size_t)OUT_DIM * IN_DIM * 2 * sizeof(_Float16));

  const int M = in_sizes[0] / IN_DIM;  // 16384

  fuse_weight_kernel<<<(OUT_DIM * IN_DIM / 8) / 256, 256, 0, stream>>>(
      weight, scale, wpert, w_hi, w_lo);
  fuse_bias_kernel<<<OUT_DIM / 256, 256, 0, stream>>>(bias, scale, bpert,
                                                      b_eff);

  dim3 grid(OUT_DIM / BN, M / BM);
  gemm_split_f16_kernel<<<grid, 256, 0, stream>>>(x, w_hi, w_lo, b_eff, out);
}
